// PointHead_59536836657347
// MI455X (gfx1250) — matrix operand order repaired and checked by reading, not compiled, not run on hardware
//
#include <hip/hip_runtime.h>
#include <hip/hip_bf16.h>

typedef __attribute__((ext_vector_type(16))) _Float16 v16h;
typedef __attribute__((ext_vector_type(8)))  _Float16 v8h;
typedef __attribute__((ext_vector_type(8)))  float    v8f;

#define B 8
#define KN 12288          // 3*4096 over-generated points per batch
#define NPTS 4096
#define NIMP 3072
#define NCOV 1024
#define MPAD 16384        // bitonic pad
#define KPAD 192          // 163 channels padded to 6*32
#define TILE_N 64
#define FSTR 200          // feat LDS stride in halves ([n][k], k contiguous)
#define HSTR 136          // h1/h2 LDS stride in halves

// ---------------- bilinear helpers (align_corners=False, zero pad) ----------

__device__ __forceinline__ float tap(const float* img, int H, int W, int x, int y) {
    if ((unsigned)x < (unsigned)W && (unsigned)y < (unsigned)H)
        return img[(size_t)y * W + x];
    return 0.0f;
}

__device__ __forceinline__ float bilin(const float* img, int H, int W, float px, float py) {
    float x0f = floorf(px), y0f = floorf(py);
    int   x0  = (int)x0f,   y0  = (int)y0f;
    float wx = px - x0f, wy = py - y0f;
    float v00 = tap(img, H, W, x0,     y0);
    float v10 = tap(img, H, W, x0 + 1, y0);
    float v01 = tap(img, H, W, x0,     y0 + 1);
    float v11 = tap(img, H, W, x0 + 1, y0 + 1);
    return v00 * (1.f - wx) * (1.f - wy) + v10 * wx * (1.f - wy)
         + v01 * (1.f - wx) * wy         + v11 * wx * wy;
}

// Precompute 4 tap offsets + zero-folded weights once per point; per-channel
// sampling then costs 4 loads + 4 FMAs with no branches.
struct BilinSetup {
    int   o00, o10, o01, o11;
    float w00, w10, w01, w11;
};

__device__ __forceinline__ BilinSetup bilin_setup(int H, int W, float px, float py) {
    float x0f = floorf(px), y0f = floorf(py);
    int   x0  = (int)x0f,   y0  = (int)y0f;
    float wx = px - x0f, wy = py - y0f;
    BilinSetup s;
    bool vx0 = (unsigned)x0 < (unsigned)W, vx1 = (unsigned)(x0 + 1) < (unsigned)W;
    bool vy0 = (unsigned)y0 < (unsigned)H, vy1 = (unsigned)(y0 + 1) < (unsigned)H;
    int cx0 = min(max(x0, 0), W - 1), cx1 = min(max(x0 + 1, 0), W - 1);
    int cy0 = min(max(y0, 0), H - 1), cy1 = min(max(y0 + 1, 0), H - 1);
    s.o00 = cy0 * W + cx0; s.o10 = cy0 * W + cx1;
    s.o01 = cy1 * W + cx0; s.o11 = cy1 * W + cx1;
    s.w00 = (vx0 && vy0) ? (1.f - wx) * (1.f - wy) : 0.f;
    s.w10 = (vx1 && vy0) ? wx * (1.f - wy)         : 0.f;
    s.w01 = (vx0 && vy1) ? (1.f - wx) * wy         : 0.f;
    s.w11 = (vx1 && vy1) ? wx * wy                 : 0.f;
    return s;
}

__device__ __forceinline__ float bilin_apply(const float* img, const BilinSetup& s) {
    return img[s.o00] * s.w00 + img[s.o10] * s.w10
         + img[s.o01] * s.w01 + img[s.o11] * s.w11;
}

// ---------------- K0: weights -> fragment-swizzled f16 ----------------------
// A-fragment layout used here, contiguous k matching load_B (16-bit A 16x32, wave32):
//   lane g = lane>>4, row m = (lane&15); element e (0..15):
//   k = kt*32 + g*16 + e
// Stored as [frag][lane][16 halves] so load_A is one contiguous 32B read/lane.
// w1f: 8 mt x 6 kt frags; w2f: 8 x 4; w3f: 1 x 4 (rows m>=3 zero, k-pad zero).

__global__ void convert_weights(const float* __restrict__ w1,
                                const float* __restrict__ w2,
                                const float* __restrict__ w3,
                                _Float16* __restrict__ w1f,
                                _Float16* __restrict__ w2f,
                                _Float16* __restrict__ w3f) {
    int t = blockIdx.x * 256 + threadIdx.x;
    const int N1 = 48 * 512, N2 = 32 * 512, N3 = 4 * 512;   // halves
    if (t < N1) {
        int f = t >> 9, within = t & 511;
        int lane = within >> 4, e = within & 15;
        int mt = f / 6, kt = f % 6, g = lane >> 4;
        int m = mt * 16 + (lane & 15);
        int k = kt * 32 + g * 16 + e;
        w1f[t] = (k < 163) ? (_Float16)w1[m * 163 + k] : (_Float16)0.0f;
    } else if (t < N1 + N2) {
        int u = t - N1;
        int f = u >> 9, within = u & 511;
        int lane = within >> 4, e = within & 15;
        int mt = f >> 2, kt = f & 3, g = lane >> 4;
        int m = mt * 16 + (lane & 15);
        int k = kt * 32 + g * 16 + e;
        w2f[u] = (_Float16)w2[m * 128 + k];
    } else if (t < N1 + N2 + N3) {
        int u = t - N1 - N2;
        int kt = u >> 9, within = u & 511;
        int lane = within >> 4, e = within & 15;
        int g = lane >> 4;
        int m = lane & 15;
        int k = kt * 32 + g * 16 + e;
        w3f[u] = (m < 3) ? (_Float16)w3[m * 128 + k] : (_Float16)0.0f;
    }
}

// ---------------- K1: uncertainty of over-generated points ------------------

__global__ void unc_kernel(const float* __restrict__ outimg,
                           const float* __restrict__ over_gen,
                           float* __restrict__ unc) {
    int gid = blockIdx.x * 256 + threadIdx.x;
    if (gid >= B * KN) return;
    int b = gid / KN;
    float x = over_gen[(size_t)gid * 2 + 0];
    float y = over_gen[(size_t)gid * 2 + 1];
    float px = x * 1024.f - 0.5f, py = y * 1024.f - 0.5f;
    const float* base = outimg + (size_t)b * 3u * 1048576u;
    BilinSetup s = bilin_setup(1024, 1024, px, py);
    float mn = 3.4e38f;
#pragma unroll
    for (int c = 0; c < 3; ++c)
        mn = fminf(mn, fabsf(bilin_apply(base + (size_t)c * 1048576u, s)));
    unc[gid] = -mn;   // top_k key (descending)
}

// ---------------- K2: per-batch top-3072 via bitonic sort in big LDS --------
// 16384 (float key, int idx) pairs = 128 KB dynamic LDS (CDNA5 WGP: 320 KB).

__global__ void topk_points(const float* __restrict__ unc,
                            const float* __restrict__ over_gen,
                            const float* __restrict__ coverage,
                            float* __restrict__ points) {
    extern __shared__ unsigned char sm[];
    float* key = (float*)sm;
    int*   sid = (int*)(sm + (size_t)MPAD * sizeof(float));
    int b = blockIdx.x;
    int tid = threadIdx.x;

    for (int i = tid; i < MPAD; i += blockDim.x) {
        if (i < KN) { key[i] = unc[(size_t)b * KN + i]; sid[i] = i; }
        else        { key[i] = -3.4e38f;                sid[i] = 0x7fffffff; }
    }
    __syncthreads();

    // bitonic: final order descending by key, ascending index on ties (== jax top_k)
    for (int k = 2; k <= MPAD; k <<= 1) {
        for (int j = k >> 1; j > 0; j >>= 1) {
            for (int i = tid; i < MPAD; i += blockDim.x) {
                int ixj = i ^ j;
                if (ixj > i) {
                    float ki = key[i], kx = key[ixj];
                    int   ii = sid[i], ix = sid[ixj];
                    bool x_before_i = (kx > ki) || (kx == ki && ix < ii);
                    bool i_before_x = (ki > kx) || (ki == kx && ii < ix);
                    bool doSwap = ((i & k) == 0) ? x_before_i : i_before_x;
                    if (doSwap) {
                        key[i] = kx; key[ixj] = ki;
                        sid[i] = ix; sid[ixj] = ii;
                    }
                }
            }
            __syncthreads();
        }
    }

    for (int j = tid; j < NPTS; j += blockDim.x) {
        float x, y;
        if (j < NIMP) {
            int s = sid[j];
            x = over_gen[((size_t)b * KN + s) * 2 + 0];
            y = over_gen[((size_t)b * KN + s) * 2 + 1];
        } else {
            int c = j - NIMP;
            x = coverage[((size_t)b * NCOV + c) * 2 + 0];
            y = coverage[((size_t)b * NCOV + c) * 2 + 1];
        }
        points[((size_t)b * NPTS + j) * 2 + 0] = x;
        points[((size_t)b * NPTS + j) * 2 + 1] = y;
    }
}

// ---------------- WMMA fragment loaders -------------------------------------

// A: fragment-swizzled global weights -> one contiguous 32B read per lane.
__device__ __forceinline__ v16h load_A(const _Float16* __restrict__ Wf,
                                       int frag, int lane) {
    return *(const v16h*)(Wf + ((size_t)(frag * 32 + lane) << 4));
}

// B 32x16 f16 from LDS laid out [n][K] (K contiguous, padded stride):
// lane reads 16 contiguous halves at column n=(lane&15)+ncol0, k = k0+(lane>>4)*16.
// 16B-aligned -> two ds_load_b128.
__device__ __forceinline__ v16h load_B(const _Float16* Bp, int kstr,
                                       int k0, int ncol0, int lane) {
    const _Float16* src = Bp + (size_t)(ncol0 + (lane & 15)) * kstr
                             + k0 + ((lane >> 4) << 4);
    v16h r;
    __builtin_memcpy(&r, src, sizeof(r));
    return r;
}

// ---------------- K3: fused gather + 3-layer MLP with WMMA ------------------

__global__ __launch_bounds__(256)
void fused_mlp(const float* __restrict__ res2, const float* __restrict__ outimg,
               const float* __restrict__ points,
               const _Float16* __restrict__ w1f, const _Float16* __restrict__ w2f,
               const _Float16* __restrict__ w3f,
               const float* __restrict__ b1, const float* __restrict__ b2,
               const float* __restrict__ b3, float* __restrict__ rend) {
    __shared__ __align__(16) _Float16 featB[TILE_N * FSTR];  // [n][k] 25.0 KB
    __shared__ __align__(16) _Float16 h1B[TILE_N * HSTR];    // [n][m] 17.0 KB
    __shared__ __align__(16) _Float16 h2B[TILE_N * HSTR];    // [n][m] 17.0 KB

    int b  = blockIdx.x >> 6;
    int n0 = (blockIdx.x & 63) * TILE_N;
    int t  = threadIdx.x;

    // warm the (L2-resident) f16 weights -> global_prefetch_b8
    __builtin_prefetch((const void*)(w1f + (size_t)(t & 63) * 384), 0, 1);

    // ---- gather: 4 channel-groups x 64 points, vector LDS stores ----
    int p   = t & 63;
    int grp = t >> 6;
    float x = points[((size_t)b * NPTS + n0 + p) * 2 + 0];
    float y = points[((size_t)b * NPTS + n0 + p) * 2 + 1];
    BilinSetup sr = bilin_setup(256, 256, x * 256.f - 0.5f, y * 256.f - 0.5f);
    BilinSetup so = bilin_setup(1024, 1024, x * 1024.f - 0.5f, y * 1024.f - 0.5f);
    const float* r2b = res2 + (size_t)b * 160u * 65536u;
    const float* otb = outimg + (size_t)b * 3u * 1048576u;
#pragma unroll
    for (int cc = 0; cc < 48; cc += 8) {
        v8h chunk;
#pragma unroll
        for (int u = 0; u < 8; ++u) {
            int c = grp * 48 + cc + u;
            float v;
            if (c < 160)      v = bilin_apply(r2b + (size_t)c * 65536u, sr);
            else if (c < 163) v = bilin_apply(otb + (size_t)(c - 160) * 1048576u, so);
            else              v = 0.0f;
            chunk[u] = (_Float16)v;
        }
        *(v8h*)(featB + (size_t)p * FSTR + grp * 48 + cc) = chunk;  // ds_store_b128
    }
    __syncthreads();

    int lane = t & 31;
    int w    = t >> 5;          // wave id, 8 waves of 32 (wave32)
    int g    = lane >> 4;
    int nl   = lane & 15;

    // ---- layer 1: [128 x 192] @ [192 x 64], relu ----
    for (int nt = 0; nt < 4; ++nt) {
        v8f acc = {};
#pragma unroll
        for (int kt = 0; kt < 6; ++kt) {
            v16h a  = load_A(w1f, w * 6 + kt, lane);
            v16h bf = load_B(featB, FSTR, kt * 32, nt * 16, lane);
            acc = __builtin_amdgcn_wmma_f32_16x16x32_f16(false, a, false, bf,
                                                         (short)0, acc, false, false);
        }
        // C/D: VGPR r holds row m = w*16 + g*8 + r at column n -> 8 contiguous
        // halves in the [n][m] layout: one ds_store_b128.
        int n = nt * 16 + nl;
        v8h hv;
#pragma unroll
        for (int r = 0; r < 8; ++r) {
            float v = acc[r] + b1[w * 16 + g * 8 + r];
            hv[r] = (_Float16)(v > 0.f ? v : 0.f);
        }
        *(v8h*)(h1B + (size_t)n * HSTR + w * 16 + g * 8) = hv;
    }
    __syncthreads();

    // ---- layer 2: [128 x 128] @ [128 x 64], relu ----
    for (int nt = 0; nt < 4; ++nt) {
        v8f acc = {};
#pragma unroll
        for (int kt = 0; kt < 4; ++kt) {
            v16h a  = load_A(w2f, w * 4 + kt, lane);
            v16h bf = load_B(h1B, HSTR, kt * 32, nt * 16, lane);
            acc = __builtin_amdgcn_wmma_f32_16x16x32_f16(false, a, false, bf,
                                                         (short)0, acc, false, false);
        }
        int n = nt * 16 + nl;
        v8h hv;
#pragma unroll
        for (int r = 0; r < 8; ++r) {
            float v = acc[r] + b2[w * 16 + g * 8 + r];
            hv[r] = (_Float16)(v > 0.f ? v : 0.f);
        }
        *(v8h*)(h2B + (size_t)n * HSTR + w * 16 + g * 8) = hv;
    }
    __syncthreads();

    // ---- layer 3: [16 x 128] @ [128 x 64] (rows 0..2 valid) ----
    if (w < 4) {
        int nt = w;
        v8f acc = {};
#pragma unroll
        for (int kt = 0; kt < 4; ++kt) {
            v16h a  = load_A(w3f, kt, lane);
            v16h bf = load_B(h2B, HSTR, kt * 32, nt * 16, lane);
            acc = __builtin_amdgcn_wmma_f32_16x16x32_f16(false, a, false, bf,
                                                         (short)0, acc, false, false);
        }
        if (g == 0) {                           // lanes 0-15 hold M rows 0..7
            int n = n0 + nt * 16 + nl;
#pragma unroll
            for (int r = 0; r < 3; ++r)
                rend[(size_t)(b * 3 + r) * NPTS + n] = acc[r] + b3[r];
        }
    }
}

// ---------------- launcher --------------------------------------------------

extern "C" void kernel_launch(void* const* d_in, const int* in_sizes, int n_in,
                              void* d_out, int out_size, void* d_ws, size_t ws_size,
                              hipStream_t stream) {
    const float* res2     = (const float*)d_in[1];
    const float* outimg   = (const float*)d_in[2];
    const float* over_gen = (const float*)d_in[3];
    const float* coverage = (const float*)d_in[4];
    const float* w1 = (const float*)d_in[5];
    const float* b1 = (const float*)d_in[6];
    const float* w2 = (const float*)d_in[7];
    const float* b2 = (const float*)d_in[8];
    const float* w3 = (const float*)d_in[9];
    const float* b3 = (const float*)d_in[10];
    float* rend = (float*)d_out;

    // workspace layout (bytes)
    char* ws = (char*)d_ws;
    float*    unc    = (float*)   (ws + 0);        // B*KN f32         = 393216 B
    float*    points = (float*)   (ws + 393216);   // B*NPTS*2 f32     = 262144 B
    _Float16* w1f    = (_Float16*)(ws + 655360);   // 48 frags * 1KB   =  49152 B
    _Float16* w2f    = (_Float16*)(ws + 704512);   // 32 frags * 1KB   =  32768 B
    _Float16* w3f    = (_Float16*)(ws + 737280);   //  4 frags * 1KB   =   4096 B

    convert_weights<<<168, 256, 0, stream>>>(w1, w2, w3, w1f, w2f, w3f);
    unc_kernel<<<(B * KN + 255) / 256, 256, 0, stream>>>(outimg, over_gen, unc);
    topk_points<<<B, 1024, MPAD * 8, stream>>>(unc, over_gen, coverage, points);
    fused_mlp<<<B * (NPTS / TILE_N), 256, 0, stream>>>(res2, outimg, points,
                                                       w1f, w2f, w3f, b1, b2, b3, rend);
}
